// MultiHeadAttention_22986664969027
// MI455X (gfx1250) — matrix operand order repaired and checked by reading, not compiled, not run on hardware
//
#include <hip/hip_runtime.h>
#include <math.h>

#define T_SEQ  2048
#define BATCH  4
#define EMBED  1024
#define HEADS  16
#define HDIM   64
#define MROWS  (T_SEQ * BATCH)   // 8192
#define S_TILE 64

typedef __bf16 bf16_t;
typedef __attribute__((ext_vector_type(16))) __bf16 v16bf;
typedef __attribute__((ext_vector_type(8)))  __bf16 v8bf;
typedef __attribute__((ext_vector_type(8)))  float  v8f;
typedef __attribute__((__vector_size__(4 * sizeof(int)))) int v4i_t;

// ---------------------------------------------------------------------------
// CDNA5 async global->LDS path (ASYNCcnt-tracked DMA); fallback is a
// synchronous load + ds_store with identical LDS layout.
// ---------------------------------------------------------------------------
#if defined(__has_builtin)
#if __has_builtin(__builtin_amdgcn_global_load_async_to_lds_b128)
#define HAVE_ASYNC_LDS 1
#endif
#if __has_builtin(__builtin_amdgcn_permlane16)
#define HAVE_PERMLANE16 1
#endif
#endif
#ifndef HAVE_ASYNC_LDS
#define HAVE_ASYNC_LDS 0
#endif
#ifndef HAVE_PERMLANE16
#define HAVE_PERMLANE16 0
#endif

__device__ __forceinline__ void copy16_g2l(const bf16_t* g, bf16_t* l) {
#if HAVE_ASYNC_LDS
  __builtin_amdgcn_global_load_async_to_lds_b128(
      (v4i_t*)(void*)g, (v4i_t*)(void*)l, 0, 0);
#else
  *(v8bf*)l = *(const v8bf*)g;
#endif
}

__device__ __forceinline__ void async_wait_all() {
#if HAVE_ASYNC_LDS
#if __has_builtin(__builtin_amdgcn_s_wait_asynccnt)
  __builtin_amdgcn_s_wait_asynccnt(0);
#else
  asm volatile("s_wait_asynccnt 0x0" ::: "memory");
#endif
#endif
}

// xor-butterfly within 16-lane halves as a VALU op (v_permlane16_b32):
// lane i of each 16-lane row reads lane i^OFF. No LDS, no wait.
template <int OFF>
__device__ __forceinline__ float half_shfl_xor(float v) {
#if HAVE_PERMLANE16
  constexpr unsigned lo = OFF == 8 ? 0xFEDCBA98u
                        : OFF == 4 ? 0x32107654u
                        : OFF == 2 ? 0x54761032u
                        :            0x67452301u;
  constexpr unsigned hi = OFF == 8 ? 0x76543210u
                        : OFF == 4 ? 0xBA98FEDCu
                        : OFF == 2 ? 0xDCFE98BAu
                        :            0xEFCDAB89u;
  int iv = __float_as_int(v);
  return __int_as_float(
      __builtin_amdgcn_permlane16(iv, iv, (int)lo, (int)hi, false, false));
#else
  return __shfl_xor(v, OFF, 16);
#endif
}

// ---------------------------------------------------------------------------
// Fragment loaders (wave32), contiguous K order; A and B use the same one.
// A (16x32 bf16, MxK): lane L -> row M=L%16; lanes 0-15 hold K 0..15,
// lanes 16-31 hold K 16..31; 2 bf16 per VGPR.
// ---------------------------------------------------------------------------
__device__ __forceinline__ v16bf frag_a(const bf16_t* base, int ld, int lane) {
  int m  = lane & 15;
  int kh = (lane >> 4) << 4;            // 0 or 16
  const bf16_t* p = base + m * ld + kh;
  v8bf lo = *(const v8bf*)(p);          // K = kh .. kh+7
  v8bf hi = *(const v8bf*)(p + 8);      // K = kh+8 .. kh+15
  v16bf r;
#pragma unroll
  for (int i = 0; i < 8; ++i) { r[i] = lo[i]; r[i + 8] = hi[i]; }
  return r;
}

// B (32x16 bf16, KxN), memory is [n][k] row-major: lane L -> column N=L%16;
// lanes 0-15 hold K=0..15, lanes 16-31 hold K=16..31 -> one 32B load/lane.
__device__ __forceinline__ v16bf frag_b(const bf16_t* base, int ld, int lane) {
  int n  = lane & 15;
  int kh = (lane >> 4) << 4;            // 0 or 16
  return *(const v16bf*)(base + n * ld + kh);
}

// C/D (16x16 f32): lane L, vgpr r -> (M = r + 8*(L/16), N = L%16)

__device__ __forceinline__ v8f zero8() {
  v8f z;
#pragma unroll
  for (int i = 0; i < 8; ++i) z[i] = 0.0f;
  return z;
}

// ---------------------------------------------------------------------------
// f32 -> bf16 conversion (4 elems / thread)
// ---------------------------------------------------------------------------
__global__ void cvt_f32_bf16_kernel(const float* __restrict__ src,
                                    bf16_t* __restrict__ dst, int n) {
  int i = (blockIdx.x * blockDim.x + threadIdx.x) * 4;
  if (i + 3 < n) {
    float4 v = *(const float4*)(src + i);
    dst[i + 0] = (bf16_t)v.x;
    dst[i + 1] = (bf16_t)v.y;
    dst[i + 2] = (bf16_t)v.z;
    dst[i + 3] = (bf16_t)v.w;
  }
}

// ---------------------------------------------------------------------------
// QKV projection: out[m,n] = sum_k X[m,k] * W[n,k] + b[n]
// block = 256 thr (8 waves), tile 128x128; wave (wm 0..1, wn 0..3) owns
// 64 rows x 32 cols = 4x2 WMMA accumulators. WHICH resolved at compile time:
// 0 = q (scaled, [bh,t,d]), 1 = k ([bh,t,d]), 2 = v (transposed [bh,d,t]).
// ---------------------------------------------------------------------------
template <int WHICH>
__global__ void qkv_proj_kernel(const bf16_t* __restrict__ X,
                                const bf16_t* __restrict__ W,
                                const float* __restrict__ bias,
                                bf16_t* __restrict__ Out,
                                float scaling) {
  const int lane = threadIdx.x & 31;
  const int wave = threadIdx.x >> 5;
  const int wm = wave >> 2;                       // 0..1
  const int wn = wave & 3;                        // 0..3
  const int m0 = blockIdx.y * 128 + wm * 64;
  const int n0 = blockIdx.x * 128 + wn * 32;

  v8f acc[4][2];
#pragma unroll
  for (int i = 0; i < 4; ++i)
#pragma unroll
    for (int j = 0; j < 2; ++j) acc[i][j] = zero8();

  for (int k0 = 0; k0 < EMBED; k0 += 32) {
    v16bf a[4];
#pragma unroll
    for (int i = 0; i < 4; ++i)
      a[i] = frag_a(X + (size_t)(m0 + i * 16) * EMBED + k0, EMBED, lane);
#pragma unroll
    for (int j = 0; j < 2; ++j) {
      v16bf b = frag_b(W + (size_t)(n0 + j * 16) * EMBED + k0, EMBED, lane);
#pragma unroll
      for (int i = 0; i < 4; ++i)
        acc[i][j] = __builtin_amdgcn_wmma_f32_16x16x32_bf16(
            false, a[i], false, b, (short)0, acc[i][j], false, false);
    }
  }

  const int halfw = lane >> 4;
  const int nn = lane & 15;
#pragma unroll
  for (int i = 0; i < 4; ++i) {
#pragma unroll
    for (int j = 0; j < 2; ++j) {
      int n = n0 + j * 16 + nn;
      float bn = bias[n];
      int h = n >> 6;            // head
      int d = n & 63;            // dim in head
#pragma unroll
      for (int r = 0; r < 8; ++r) {
        int m = m0 + i * 16 + r + 8 * halfw;    // m = t*BATCH + b
        int t = m >> 2;
        int bb = m & 3;
        int bh = bb * HEADS + h;
        float c = acc[i][j][r] + bn;
        if (WHICH == 0) {
          c *= scaling;
          Out[((size_t)bh * T_SEQ + t) * HDIM + d] = (bf16_t)c;
        } else if (WHICH == 1) {
          Out[((size_t)bh * T_SEQ + t) * HDIM + d] = (bf16_t)c;
        } else {
          Out[((size_t)bh * HDIM + d) * T_SEQ + t] = (bf16_t)c;   // transposed
        }
      }
    }
  }
}

// ---------------------------------------------------------------------------
// Flash attention: grid (T/128, B*H), block 256 (8 waves), 1 block/CU so the
// compiler keeps Q fragments + accumulators resident (no spills).
// K/V 64x64 tiles cooperatively staged into LDS via async DMA,
// double-buffered so the next tile's copy overlaps this tile's WMMA work.
// Row sums of P come from an extra WMMA against an all-ones B matrix
// (row sum replicated across all columns/lanes); row max uses VALU
// permlane16 butterflies (no LDS traffic).
// ---------------------------------------------------------------------------
__global__ void __launch_bounds__(256, 1)
flash_attn_kernel(const bf16_t* __restrict__ Q,
                  const bf16_t* __restrict__ K,
                  const bf16_t* __restrict__ Vt,
                  const float* __restrict__ mask,
                  bf16_t* __restrict__ Aout) {
  __shared__ bf16_t kbuf[2][S_TILE * HDIM];   // 2 x 8KB, rows s, ld=HDIM
  __shared__ bf16_t vbuf[2][HDIM * S_TILE];   // 2 x 8KB, rows d, ld=S_TILE
  __shared__ bf16_t pbuf[8][16 * S_TILE];     // 16KB per-wave P staging

  const int tid  = threadIdx.x;
  const int lane = tid & 31;
  const int wave = tid >> 5;
  const int bh = blockIdx.y;
  const int trow0 = blockIdx.x * 128 + wave * 16;

  const bf16_t* Qb = Q  + (size_t)bh * T_SEQ * HDIM;
  const bf16_t* Kb = K  + (size_t)bh * T_SEQ * HDIM;
  const bf16_t* Vb = Vt + (size_t)bh * HDIM * T_SEQ;

  // Cooperative tile staging: 8KB tile = 512 16B chunks, 2 per thread.
  auto issue_tile = [&](int s0, int p) {
#pragma unroll
    for (int c = tid; c < 512; c += 256)      // K tile: contiguous block
      copy16_g2l(Kb + (size_t)s0 * HDIM + c * 8, &kbuf[p][c * 8]);
#pragma unroll
    for (int c = tid; c < 512; c += 256) {    // V tile: 64 strided rows
      int row = c >> 3;
      int col = (c & 7) * 8;
      copy16_g2l(Vb + (size_t)row * T_SEQ + s0 + col, &vbuf[p][row * S_TILE + col]);
    }
  };

  // Q fragments for this wave's 16 rows (K = 0..63), kept resident
  v16bf aq[2];
  aq[0] = frag_a(Qb + (size_t)trow0 * HDIM + 0,  HDIM, lane);
  aq[1] = frag_a(Qb + (size_t)trow0 * HDIM + 32, HDIM, lane);

  // All-ones B fragment: P x ones = row sums replicated across columns.
  v16bf ones;
#pragma unroll
  for (int i = 0; i < 16; ++i) ones[i] = (bf16_t)1.0f;

  v8f acc[4];
#pragma unroll
  for (int dj = 0; dj < 4; ++dj) acc[dj] = zero8();

  float mrun[8], lrun[8];
#pragma unroll
  for (int r = 0; r < 8; ++r) { mrun[r] = -INFINITY; lrun[r] = 0.0f; }

  const int halfw = lane >> 4;
  const int nn = lane & 15;
  bf16_t* lbuf = &pbuf[wave][0];

  issue_tile(0, 0);

  for (int it = 0; it < T_SEQ / S_TILE; ++it) {
    const int s0 = it * S_TILE;
    const int p = it & 1;

    async_wait_all();          // this wave's copies for tile `it` landed
    __syncthreads();           // every wave's copies landed
    if (s0 + S_TILE < T_SEQ) issue_tile(s0 + S_TILE, p ^ 1);

    // ---- batched mask loads (independent -> clause-able) ----
    float mv[4][8];
#pragma unroll
    for (int r = 0; r < 8; ++r) {
      const float* mrow =
          mask + (size_t)(trow0 + r + 8 * halfw) * T_SEQ + s0 + nn;
#pragma unroll
      for (int j = 0; j < 4; ++j) mv[j][r] = mrow[j * 16];
    }

    // ---- scores S = q . k^T  (16 x 64), K tile from LDS ----
    v8f s[4];
#pragma unroll
    for (int j = 0; j < 4; ++j) {
      s[j] = zero8();
#pragma unroll
      for (int kk = 0; kk < 2; ++kk) {
        v16bf bk = frag_b(&kbuf[p][(j * 16) * HDIM + kk * 32], HDIM, lane);
        s[j] = __builtin_amdgcn_wmma_f32_16x16x32_bf16(
            false, aq[kk], false, bk, (short)0, s[j], false, false);
      }
    }
    // ---- add mask ----
#pragma unroll
    for (int j = 0; j < 4; ++j)
#pragma unroll
      for (int r = 0; r < 8; ++r) s[j][r] += mv[j][r];

    // ---- row max: VALU permlane16 butterfly over 16-lane halves ----
    float vmax[8];
#pragma unroll
    for (int r = 0; r < 8; ++r) {
      float v = fmaxf(fmaxf(s[0][r], s[1][r]), fmaxf(s[2][r], s[3][r]));
      v = fmaxf(v, half_shfl_xor<8>(v));
      v = fmaxf(v, half_shfl_xor<4>(v));
      v = fmaxf(v, half_shfl_xor<2>(v));
      v = fmaxf(v, half_shfl_xor<1>(v));
      vmax[r] = v;
    }

    float mnew[8], alpha[8];
#pragma unroll
    for (int r = 0; r < 8; ++r) {
      mnew[r] = fmaxf(mrun[r], vmax[r]);
      alpha[r] = __expf(mrun[r] - mnew[r]);
      mrun[r] = mnew[r];
    }

    // ---- exponentiate ----
#pragma unroll
    for (int j = 0; j < 4; ++j)
#pragma unroll
      for (int r = 0; r < 8; ++r) s[j][r] = __expf(s[j][r] - mnew[r]);

    // ---- rescale O ----
#pragma unroll
    for (int dj = 0; dj < 4; ++dj)
#pragma unroll
      for (int r = 0; r < 8; ++r) acc[dj][r] *= alpha[r];

    // ---- P (bf16) -> per-wave LDS row-major, then back as A-fragments ----
#pragma unroll
    for (int j = 0; j < 4; ++j)
#pragma unroll
      for (int r = 0; r < 8; ++r)
        lbuf[(r + 8 * halfw) * S_TILE + j * 16 + nn] = (bf16_t)s[j][r];
    // DS ops from the same wave complete in order; no barrier needed here.

    v8f psum = zero8();        // row sums of P via WMMA against ones
#pragma unroll
    for (int kk = 0; kk < 2; ++kk) {
      v16bf ap = frag_a(lbuf + kk * 32, S_TILE, lane);
      psum = __builtin_amdgcn_wmma_f32_16x16x32_bf16(
          false, ap, false, ones, (short)0, psum, false, false);
#pragma unroll
      for (int dj = 0; dj < 4; ++dj) {
        v16bf bv = frag_b(&vbuf[p][(dj * 16) * S_TILE + kk * 32], S_TILE, lane);
        acc[dj] = __builtin_amdgcn_wmma_f32_16x16x32_bf16(
            false, ap, false, bv, (short)0, acc[dj], false, false);
      }
    }
#pragma unroll
    for (int r = 0; r < 8; ++r) lrun[r] = lrun[r] * alpha[r] + psum[r];

    __syncthreads();           // all waves done reading buffers [p]
  }

  // ---- normalize + store to [t*B+b, h*64+d] bf16 ----
  const int b = bh >> 4;         // BATCH index (H == 16)
  const int h = bh & 15;
  float inv[8];
#pragma unroll
  for (int r = 0; r < 8; ++r) inv[r] = 1.0f / lrun[r];
#pragma unroll
  for (int dj = 0; dj < 4; ++dj) {
#pragma unroll
    for (int r = 0; r < 8; ++r) {
      int t = trow0 + r + 8 * halfw;
      int col = h * HDIM + dj * 16 + nn;
      Aout[(size_t)(t * BATCH + b) * EMBED + col] = (bf16_t)(acc[dj][r] * inv[r]);
    }
  }
}

// ---------------------------------------------------------------------------
// Output projection: out[m,n] = sum_k A[m,k] * Wo[n,k] + bo[n], f32 store.
// ---------------------------------------------------------------------------
__global__ void out_proj_kernel(const bf16_t* __restrict__ A,
                                const bf16_t* __restrict__ Wo16,
                                const float* __restrict__ bo,
                                float* __restrict__ out) {
  const int lane = threadIdx.x & 31;
  const int wave = threadIdx.x >> 5;
  const int wm = wave >> 2;
  const int wn = wave & 3;
  const int m0 = blockIdx.y * 128 + wm * 64;
  const int n0 = blockIdx.x * 128 + wn * 32;

  v8f acc[4][2];
#pragma unroll
  for (int i = 0; i < 4; ++i)
#pragma unroll
    for (int j = 0; j < 2; ++j) acc[i][j] = zero8();

  for (int k0 = 0; k0 < EMBED; k0 += 32) {
    v16bf a[4];
#pragma unroll
    for (int i = 0; i < 4; ++i)
      a[i] = frag_a(A + (size_t)(m0 + i * 16) * EMBED + k0, EMBED, lane);
#pragma unroll
    for (int j = 0; j < 2; ++j) {
      v16bf b = frag_b(Wo16 + (size_t)(n0 + j * 16) * EMBED + k0, EMBED, lane);
#pragma unroll
      for (int i = 0; i < 4; ++i)
        acc[i][j] = __builtin_amdgcn_wmma_f32_16x16x32_bf16(
            false, a[i], false, b, (short)0, acc[i][j], false, false);
    }
  }

  const int halfw = lane >> 4;
  const int nn = lane & 15;
#pragma unroll
  for (int i = 0; i < 4; ++i) {
#pragma unroll
    for (int j = 0; j < 2; ++j) {
      int n = n0 + j * 16 + nn;
      float bn = bo[n];
#pragma unroll
      for (int r = 0; r < 8; ++r) {
        int m = m0 + i * 16 + r + 8 * halfw;
        out[(size_t)m * EMBED + n] = acc[i][j][r] + bn;
      }
    }
  }
}

// ---------------------------------------------------------------------------
extern "C" void kernel_launch(void* const* d_in, const int* in_sizes, int n_in,
                              void* d_out, int out_size, void* d_ws, size_t ws_size,
                              hipStream_t stream) {
  (void)in_sizes; (void)n_in; (void)out_size;
  const float* query = (const float*)d_in[0];
  const float* mask  = (const float*)d_in[1];
  const float* Wq = (const float*)d_in[2];
  const float* bq = (const float*)d_in[3];
  const float* Wk = (const float*)d_in[4];
  const float* bk = (const float*)d_in[5];
  const float* Wv = (const float*)d_in[6];
  const float* bv = (const float*)d_in[7];
  const float* Wo = (const float*)d_in[8];
  const float* bo = (const float*)d_in[9];
  float* out = (float*)d_out;

  char* ws = (char*)d_ws;
  size_t off = 0;
  const size_t xBytes = (size_t)MROWS * EMBED * sizeof(bf16_t);
  const size_t wBytes = (size_t)EMBED * EMBED * sizeof(bf16_t);
  const size_t hBytes = (size_t)BATCH * HEADS * T_SEQ * HDIM * sizeof(bf16_t);
  bf16_t* X16  = (bf16_t*)(ws + off); off += xBytes;
  bf16_t* Wq16 = (bf16_t*)(ws + off); off += wBytes;
  bf16_t* Wk16 = (bf16_t*)(ws + off); off += wBytes;
  bf16_t* Wv16 = (bf16_t*)(ws + off); off += wBytes;
  bf16_t* Wo16 = (bf16_t*)(ws + off); off += wBytes;
  bf16_t* Q16  = (bf16_t*)(ws + off); off += hBytes;
  bf16_t* K16  = (bf16_t*)(ws + off); off += hBytes;
  bf16_t* Vt16 = (bf16_t*)(ws + off); off += hBytes;
  bf16_t* A16  = (bf16_t*)(ws + off); off += xBytes;
  if (ws_size < off) return;   // ~88 MB required

  // Stage 1: f32 -> bf16
  {
    int nx = MROWS * EMBED;
    int nw = EMBED * EMBED;
    cvt_f32_bf16_kernel<<<nx / 4 / 256, 256, 0, stream>>>(query, X16, nx);
    cvt_f32_bf16_kernel<<<nw / 4 / 256, 256, 0, stream>>>(Wq, Wq16, nw);
    cvt_f32_bf16_kernel<<<nw / 4 / 256, 256, 0, stream>>>(Wk, Wk16, nw);
    cvt_f32_bf16_kernel<<<nw / 4 / 256, 256, 0, stream>>>(Wv, Wv16, nw);
    cvt_f32_bf16_kernel<<<nw / 4 / 256, 256, 0, stream>>>(Wo, Wo16, nw);
  }

  // Stage 2: QKV projections (WHICH resolved at compile time)
  {
    dim3 grid(EMBED / 128, MROWS / 128);
    float scaling = 1.0f / sqrtf((float)HDIM);
    qkv_proj_kernel<0><<<grid, 256, 0, stream>>>(X16, Wq16, bq, Q16,  scaling);
    qkv_proj_kernel<1><<<grid, 256, 0, stream>>>(X16, Wk16, bk, K16,  scaling);
    qkv_proj_kernel<2><<<grid, 256, 0, stream>>>(X16, Wv16, bv, Vt16, scaling);
  }

  // Stage 3: flash attention
  {
    dim3 grid(T_SEQ / 128, BATCH * HEADS);
    flash_attn_kernel<<<grid, 256, 0, stream>>>(Q16, K16, Vt16, mask, A16);
  }

  // Stage 4: output projection
  {
    dim3 grid(EMBED / 128, MROWS / 128);
    out_proj_kernel<<<grid, 256, 0, stream>>>(A16, Wo16, bo, out);
  }
}
